// SSMCompositionBlock_64596308132158
// MI455X (gfx1250) — matrix operand order repaired and checked by reading, compiled, not run on hardware
//
#include <hip/hip_runtime.h>
#include <hip/hip_bf16.h>

// ---------------------------------------------------------------------------
// Types for CDNA5 WMMA (wave32, v_wmma_f32_16x16x32_bf16)
// ---------------------------------------------------------------------------
typedef __attribute__((ext_vector_type(16))) __bf16 v16bf;
typedef __attribute__((ext_vector_type(8)))  __bf16 v8bf;
typedef __attribute__((ext_vector_type(8)))  float  v8f;

union BF16x16 { v16bf v; v8bf h[2]; };

// A-fragment (16x32 bf16, MxK): lane m = lane&15, half = lane>>4.
// element e -> k = (e<8 ? half*8+e : 16+half*8+(e-8)).
// Two aligned 16B LDS loads per fragment.
__device__ __forceinline__ v16bf frag_a(const __bf16* row, int half) {
    BF16x16 u;
    u.h[0] = *reinterpret_cast<const v8bf*>(row + half * 8);
    u.h[1] = *reinterpret_cast<const v8bf*>(row + 16 + half * 8);
    return u.v;
}
// B-fragment (32x16 bf16, KxN): lane n = lane&15, half = lane>>4.
// element e -> k = (e<8 ? half*8+e : 16+half*8+(e-8)). Two aligned 16B LDS loads.
__device__ __forceinline__ v16bf frag_b(const __bf16* row, int half) {
    BF16x16 u;
    u.h[0] = *reinterpret_cast<const v8bf*>(row + half * 8);
    u.h[1] = *reinterpret_cast<const v8bf*>(row + 16 + half * 8);
    return u.v;
}

__device__ __forceinline__ v8f wmma_bf16(v16bf a, v16bf b, v8f c) {
    return __builtin_amdgcn_wmma_f32_16x16x32_bf16(false, a, false, b, (short)0, c,
                                                   false, false);
}

// ---------------------------------------------------------------------------
// LayerNorm: one block (256 thr) per row of H columns.
// ---------------------------------------------------------------------------
__global__ __launch_bounds__(256)
void layernorm_kernel(const float* __restrict__ x, const float* __restrict__ g,
                      const float* __restrict__ beta, float* __restrict__ y, int H)
{
    const int row = blockIdx.x;
    const float* xr = x + (size_t)row * H;
    float s = 0.f, s2 = 0.f;
    for (int i = threadIdx.x; i < H; i += 256) {
        float v = xr[i];
        s += v; s2 += v * v;
    }
    for (int off = 16; off > 0; off >>= 1) {
        s  += __shfl_xor(s,  off);
        s2 += __shfl_xor(s2, off);
    }
    __shared__ float redS[8], redS2[8], mv[2];
    const int wave = threadIdx.x >> 5, lane = threadIdx.x & 31;
    if (lane == 0) { redS[wave] = s; redS2[wave] = s2; }
    __syncthreads();
    if (threadIdx.x == 0) {
        float a = 0.f, b2 = 0.f;
        for (int i = 0; i < 8; ++i) { a += redS[i]; b2 += redS2[i]; }
        float mu = a / H;
        mv[0] = mu;
        mv[1] = rsqrtf(b2 / H - mu * mu + 1e-5f);
    }
    __syncthreads();
    const float mu = mv[0], inv = mv[1];
    float* yr = y + (size_t)row * H;
    for (int i = threadIdx.x; i < H; i += 256)
        yr[i] = (xr[i] - mu) * inv * g[i] + beta[i];
}

// ---------------------------------------------------------------------------
// Generic WMMA GEMM: C[M,N] = A[M,K] @ W^T (+bias)(+residual)(+mod)
//   b_is_nk==1: W stored [N,K] (weight matrices, C = A @ W^T)
//   b_is_nk==0: W stored [K,N] (plain matmul, C = A @ W)
//   nsplit: blocks with nbase >= nsplit read activations from A1 (QKV fusion)
//   modv:  per-batch [B,N] broadcast add; batch index = row / Tmod
// Tile: 128x64 per block, 8 waves -> 32x32 per wave, K step 32.
// ---------------------------------------------------------------------------
__global__ __launch_bounds__(256)
void gemm_kernel(const float* __restrict__ A0, const float* __restrict__ A1, int nsplit,
                 const float* __restrict__ W, int b_is_nk,
                 const float* __restrict__ bias, const float* __restrict__ residual,
                 const float* __restrict__ modv, int Tmod,
                 float* __restrict__ C, int Msz, int Nsz, int Ksz)
{
    __shared__ __bf16 As[128][32];   // 8 KB
    __shared__ __bf16 Bs[64][32];    // 4 KB

    const int tid  = threadIdx.x;
    const int wave = tid >> 5, lane = tid & 31;
    const int wm = wave & 3, wn = wave >> 2;       // 4x2 wave grid
    const int half = lane >> 4, l16 = lane & 15;
    const int mbase = blockIdx.x * 128, nbase = blockIdx.y * 64;
    const float* Ap = (nbase < nsplit) ? A0 : A1;

    v8f acc[2][2];
    #pragma unroll
    for (int i = 0; i < 2; ++i)
        #pragma unroll
        for (int j = 0; j < 2; ++j)
            #pragma unroll
            for (int e = 0; e < 8; ++e) acc[i][j][e] = 0.f;

    for (int kb = 0; kb < Ksz; kb += 32) {
        // --- A tile (128x32 f32 -> bf16), 4 float4 per thread
        #pragma unroll
        for (int it = 0; it < 4; ++it) {
            int idx = tid + it * 256;              // 0..1023 float4 slots
            int r = idx >> 3, c4 = (idx & 7) << 2;
            const float4 f = *reinterpret_cast<const float4*>(
                Ap + (size_t)(mbase + r) * Ksz + kb + c4);
            As[r][c4 + 0] = (__bf16)f.x; As[r][c4 + 1] = (__bf16)f.y;
            As[r][c4 + 2] = (__bf16)f.z; As[r][c4 + 3] = (__bf16)f.w;
        }
        // --- B tile (64x32 bf16, stored [n][k])
        if (b_is_nk) {
            #pragma unroll
            for (int it = 0; it < 2; ++it) {
                int idx = tid + it * 256;          // 0..511
                int r = idx >> 3, c4 = (idx & 7) << 2;
                const float4 f = *reinterpret_cast<const float4*>(
                    W + (size_t)(nbase + r) * Ksz + kb + c4);
                Bs[r][c4 + 0] = (__bf16)f.x; Bs[r][c4 + 1] = (__bf16)f.y;
                Bs[r][c4 + 2] = (__bf16)f.z; Bs[r][c4 + 3] = (__bf16)f.w;
            }
        } else {
            #pragma unroll
            for (int it = 0; it < 2; ++it) {
                int idx = tid + it * 256;          // 0..511
                int kk = idx >> 4, c4 = (idx & 15) << 2;
                const float4 f = *reinterpret_cast<const float4*>(
                    W + (size_t)(kb + kk) * Nsz + nbase + c4);
                Bs[c4 + 0][kk] = (__bf16)f.x; Bs[c4 + 1][kk] = (__bf16)f.y;
                Bs[c4 + 2][kk] = (__bf16)f.z; Bs[c4 + 3][kk] = (__bf16)f.w;
            }
        }
        __syncthreads();

        v16bf afr[2], bfr[2];
        afr[0] = frag_a(&As[wm * 32 + l16][0], half);
        afr[1] = frag_a(&As[wm * 32 + 16 + l16][0], half);
        bfr[0] = frag_b(&Bs[wn * 32 + l16][0], half);
        bfr[1] = frag_b(&Bs[wn * 32 + 16 + l16][0], half);
        #pragma unroll
        for (int mt = 0; mt < 2; ++mt)
            #pragma unroll
            for (int nt = 0; nt < 2; ++nt)
                acc[mt][nt] = wmma_bf16(afr[mt], bfr[nt], acc[mt][nt]);
        __syncthreads();
    }

    // --- epilogue (C/D layout: row = half*8+j, col = l16)
    #pragma unroll
    for (int mt = 0; mt < 2; ++mt) {
        #pragma unroll
        for (int nt = 0; nt < 2; ++nt) {
            const int col = nbase + wn * 32 + nt * 16 + l16;
            #pragma unroll
            for (int j = 0; j < 8; ++j) {
                const int row = mbase + wm * 32 + mt * 16 + half * 8 + j;
                float v = acc[mt][nt][j];
                if (bias)     v += bias[col];
                if (residual) v += residual[(size_t)row * Nsz + col];
                if (modv)     v += modv[(size_t)(row / Tmod) * Nsz + col];
                C[(size_t)row * Nsz + col] = v;
            }
        }
    }
}

// ---------------------------------------------------------------------------
// Flash-attention (non-causal), NH=8, HD=128, T=2048.
// grid = (T/64 q-tiles, B*NH). 256 threads = 8 waves.
// qkv layout: [B*T, 3072] with Q at +0, K at +1024, V at +2048 (per-head *128).
// Wave w: S rows (w&3)*16 (16), S cols (w>>2)*32 (32);
//         O rows (w&3)*16, O cols (w>>2)*64 -> same row<->lane map => in-reg softmax.
// ---------------------------------------------------------------------------
__global__ __launch_bounds__(256)
void attn_kernel(const float* __restrict__ qkv, float* __restrict__ out)
{
    constexpr int QKVW = 3072, Tt = 2048;
    constexpr float SCALE = 0.08838834764831845f;   // 1/sqrt(128)

    const int qt = blockIdx.x;          // 0..31
    const int bh = blockIdx.y;          // 0..31
    const int b = bh >> 3, h = bh & 7;
    const int qbase = qt * 64;

    __shared__ __bf16 Qs[64][128];      // 16 KB
    __shared__ __bf16 Ks[64][128];      // 16 KB (first 8 KB re-used as Ps)
    __shared__ __bf16 Vt[128][64];      // 16 KB, transposed [d][key]
    __shared__ float  wmaxs[2][64];
    __shared__ float  wsums[2][64];
    __bf16 (*Ps)[64] = reinterpret_cast<__bf16(*)[64]>(&Ks[0][0]);

    const int tid = threadIdx.x, wave = tid >> 5, lane = tid & 31;
    const int half = lane >> 4, l16 = lane & 15;
    const int wr = wave & 3, wn = wave >> 2;
    const int rbase = wr * 16;
    const size_t base_m = (size_t)b * Tt;

    // --- load Q tile (64x128), f32 -> bf16
    #pragma unroll
    for (int it = 0; it < 8; ++it) {
        int idx = tid + it * 256;       // 0..2047 float4 slots
        int r = idx >> 5, d4 = (idx & 31) << 2;
        const float4 f = *reinterpret_cast<const float4*>(
            qkv + (base_m + qbase + r) * QKVW + (size_t)h * 128 + d4);
        Qs[r][d4 + 0] = (__bf16)f.x; Qs[r][d4 + 1] = (__bf16)f.y;
        Qs[r][d4 + 2] = (__bf16)f.z; Qs[r][d4 + 3] = (__bf16)f.w;
    }
    __syncthreads();

    v16bf qfr[4];
    #pragma unroll
    for (int kk = 0; kk < 4; ++kk)
        qfr[kk] = frag_a(&Qs[rbase + l16][kk * 32], half);

    float mprev[8], lprev[8];
    v8f oacc[4];
    #pragma unroll
    for (int j = 0; j < 8; ++j) { mprev[j] = -1e30f; lprev[j] = 0.f; }
    #pragma unroll
    for (int nt = 0; nt < 4; ++nt)
        #pragma unroll
        for (int j = 0; j < 8; ++j) oacc[nt][j] = 0.f;

    for (int kc = 0; kc < 32; ++kc) {
        __syncthreads();                         // prev-iter Ps/Vt readers done
        // --- load K chunk -> Ks[key][d], V chunk -> Vt[d][key]
        #pragma unroll
        for (int it = 0; it < 8; ++it) {
            int idx = tid + it * 256;
            int r = idx >> 5, d4 = (idx & 31) << 2;
            size_t src = (base_m + (size_t)kc * 64 + r) * QKVW + 1024 +
                         (size_t)h * 128 + d4;
            const float4 fk = *reinterpret_cast<const float4*>(qkv + src);
            Ks[r][d4 + 0] = (__bf16)fk.x; Ks[r][d4 + 1] = (__bf16)fk.y;
            Ks[r][d4 + 2] = (__bf16)fk.z; Ks[r][d4 + 3] = (__bf16)fk.w;
            const float4 fv = *reinterpret_cast<const float4*>(qkv + src + 1024);
            Vt[d4 + 0][r] = (__bf16)fv.x; Vt[d4 + 1][r] = (__bf16)fv.y;
            Vt[d4 + 2][r] = (__bf16)fv.z; Vt[d4 + 3][r] = (__bf16)fv.w;
        }
        __syncthreads();

        // --- S = Q K^T (16x32 per wave)
        v8f sacc[2];
        #pragma unroll
        for (int nt = 0; nt < 2; ++nt)
            #pragma unroll
            for (int j = 0; j < 8; ++j) sacc[nt][j] = 0.f;
        #pragma unroll
        for (int kk = 0; kk < 4; ++kk) {
            v16bf b0 = frag_b(&Ks[wn * 32 + l16][kk * 32], half);
            v16bf b1 = frag_b(&Ks[wn * 32 + 16 + l16][kk * 32], half);
            sacc[0] = wmma_bf16(qfr[kk], b0, sacc[0]);
            sacc[1] = wmma_bf16(qfr[kk], b1, sacc[1]);
        }
        __syncthreads();                         // all Ks reads done

        // --- online softmax, fragment-resident
        float mx[8];
        #pragma unroll
        for (int j = 0; j < 8; ++j) {
            sacc[0][j] *= SCALE; sacc[1][j] *= SCALE;
            mx[j] = fmaxf(sacc[0][j], sacc[1][j]);
        }
        #pragma unroll
        for (int off = 1; off < 16; off <<= 1)
            #pragma unroll
            for (int j = 0; j < 8; ++j)
                mx[j] = fmaxf(mx[j], __shfl_xor(mx[j], off));
        if (l16 == 0) {
            #pragma unroll
            for (int j = 0; j < 8; ++j) wmaxs[wn][rbase + half * 8 + j] = mx[j];
        }
        __syncthreads();

        float mnew[8], alpha[8], ls[8];
        #pragma unroll
        for (int j = 0; j < 8; ++j) {
            const int r = rbase + half * 8 + j;
            const float mc = fmaxf(wmaxs[0][r], wmaxs[1][r]);
            mnew[j]  = fmaxf(mprev[j], mc);
            alpha[j] = __expf(mprev[j] - mnew[j]);
        }
        #pragma unroll
        for (int j = 0; j < 8; ++j) {
            const float p0 = __expf(sacc[0][j] - mnew[j]);
            const float p1 = __expf(sacc[1][j] - mnew[j]);
            const int r = rbase + half * 8 + j;
            Ps[r][wn * 32 + l16]      = (__bf16)p0;
            Ps[r][wn * 32 + 16 + l16] = (__bf16)p1;
            ls[j] = p0 + p1;
        }
        #pragma unroll
        for (int off = 1; off < 16; off <<= 1)
            #pragma unroll
            for (int j = 0; j < 8; ++j)
                ls[j] += __shfl_xor(ls[j], off);
        if (l16 == 0) {
            #pragma unroll
            for (int j = 0; j < 8; ++j) wsums[wn][rbase + half * 8 + j] = ls[j];
        }
        __syncthreads();                         // Ps complete, sums complete
        #pragma unroll
        for (int j = 0; j < 8; ++j) {
            const int r = rbase + half * 8 + j;
            lprev[j] = alpha[j] * lprev[j] + wsums[0][r] + wsums[1][r];
            mprev[j] = mnew[j];
        }
        #pragma unroll
        for (int nt = 0; nt < 4; ++nt)
            #pragma unroll
            for (int j = 0; j < 8; ++j) oacc[nt][j] *= alpha[j];

        // --- O += P @ V  (16x64 per wave)
        #pragma unroll
        for (int k2 = 0; k2 < 2; ++k2) {
            v16bf pa = frag_a(&Ps[rbase + l16][k2 * 32], half);
            #pragma unroll
            for (int nt = 0; nt < 4; ++nt) {
                v16bf vb = frag_b(&Vt[wn * 64 + nt * 16 + l16][k2 * 32], half);
                oacc[nt] = wmma_bf16(pa, vb, oacc[nt]);
            }
        }
    }

    // --- normalize & store: out[(b*T + q), h*128 + d]
    #pragma unroll
    for (int nt = 0; nt < 4; ++nt) {
        const int col = h * 128 + wn * 64 + nt * 16 + l16;
        #pragma unroll
        for (int j = 0; j < 8; ++j) {
            const int r = rbase + half * 8 + j;
            const size_t m = base_m + qbase + r;
            out[m * 1024 + col] = oacc[nt][j] / lprev[j];
        }
    }
}

// ---------------------------------------------------------------------------
// Sequential SSM scan: s = silu(s @ A^T) + u_t ; one block, 256 threads.
// U/states layout: [B*T, 64] (row m = b*T + t).
// ---------------------------------------------------------------------------
__global__ __launch_bounds__(256)
void scan_kernel(const float* __restrict__ A, const float* __restrict__ U,
                 float* __restrict__ states, int T)
{
    __shared__ float As[64][64];    // A[j][k]
    __shared__ float sbuf[4][64];
    const int tid = threadIdx.x;
    for (int i = tid; i < 64 * 64; i += 256) As[i >> 6][i & 63] = A[i];
    const int b = tid >> 6, j = tid & 63;
    sbuf[b][j] = 0.f;
    __syncthreads();
    for (int t = 0; t < T; ++t) {
        float acc = 0.f;
        #pragma unroll 8
        for (int k = 0; k < 64; ++k) acc += As[j][k] * sbuf[b][k];
        const float sv = acc / (1.f + __expf(-acc));     // silu
        const float sn = sv + U[((size_t)b * T + t) * 64 + j];
        __syncthreads();
        sbuf[b][j] = sn;
        states[((size_t)b * T + t) * 64 + j] = sn;
        __syncthreads();
    }
}

// ---------------------------------------------------------------------------
// program_probs mean over T: [B,T,P] -> [B,P]   (B*P = 256)
// ---------------------------------------------------------------------------
__global__ __launch_bounds__(256)
void meanprog_kernel(const float* __restrict__ pp, float* __restrict__ pm,
                     int T, int P)
{
    const int idx = threadIdx.x;
    const int b = idx / P, p = idx % P;
    float s = 0.f;
    for (int t = 0; t < T; ++t) s += pp[((size_t)b * T + t) * P + p];
    pm[idx] = s / (float)T;
}

// mod[b,h] = pm[b,:] @ W_prog[h,:] + b_prog[h]
__global__ __launch_bounds__(256)
void mod_kernel(const float* __restrict__ pm, const float* __restrict__ Wp,
                const float* __restrict__ bp, float* __restrict__ modv,
                int P, int H)
{
    const int idx = blockIdx.x * 256 + threadIdx.x;   // B*H
    const int b = idx / H, hc = idx % H;
    float s = bp[hc];
    for (int p = 0; p < P; ++p) s += pm[b * P + p] * Wp[(size_t)hc * P + p];
    modv[idx] = s;
}

// ---------------------------------------------------------------------------
// Orchestration
// ---------------------------------------------------------------------------
extern "C" void kernel_launch(void* const* d_in, const int* in_sizes, int n_in,
                              void* d_out, int out_size, void* d_ws, size_t ws_size,
                              hipStream_t stream)
{
    (void)in_sizes; (void)n_in; (void)out_size; (void)ws_size;
    constexpr int B = 4, T = 2048, H = 1024, S = 64, P = 64;
    constexpr int M = B * T;                   // 8192

    const float* builder = (const float*)d_in[0];
    const float* base    = (const float*)d_in[1];
    /* d_in[2] input_ids: unused by reference */
    const float* pp      = (const float*)d_in[3];
    const float* Amat    = (const float*)d_in[4];
    const float* Bmat    = (const float*)d_in[5];
    const float* Cmat    = (const float*)d_in[6];
    const float* W_in    = (const float*)d_in[7];
    const float* b_in    = (const float*)d_in[8];
    const float* W_qkv   = (const float*)d_in[9];
    const float* b_qkv   = (const float*)d_in[10];
    const float* W_o     = (const float*)d_in[11];
    const float* b_o     = (const float*)d_in[12];
    const float* W_prog  = (const float*)d_in[13];
    const float* b_prog  = (const float*)d_in[14];
    const float* g1      = (const float*)d_in[15];
    const float* beta1   = (const float*)d_in[16];
    const float* g2      = (const float*)d_in[17];
    const float* beta2   = (const float*)d_in[18];

    char* ws = (char*)d_ws;
    const size_t MB = 1024ull * 1024ull;
    float* xn     = (float*)(ws + 0 * MB);      // 32 MB, reused as xn2
    float* basen  = (float*)(ws + 32 * MB);     // 32 MB, reused as hidden
    float* qkvbuf = (float*)(ws + 64 * MB);     // 96 MB (freed after attention)
    float* attnb  = (float*)(ws + 160 * MB);    // 32 MB
    float* x1     = (float*)(ws + 192 * MB);    // 32 MB
    float* Ubuf   = (float*)(ws + 64 * MB);     // 2 MB  (in dead qkv region)
    float* statesb= (float*)(ws + 68 * MB);     // 2 MB
    float* pm     = (float*)(ws + 72 * MB);     // 1 KB
    float* modb   = (float*)(ws + 72 * MB + 64 * 1024);  // 16 KB
    const int BIG = 1 << 30;

    // 1) layernorms (norm1 on builder and base)
    layernorm_kernel<<<M, 256, 0, stream>>>(builder, g1, beta1, xn, H);
    layernorm_kernel<<<M, 256, 0, stream>>>(base, g1, beta1, basen, H);
    // 2) fused QKV: Q from xn, K/V from basen
    gemm_kernel<<<dim3(M / 128, (3 * H) / 64), 256, 0, stream>>>(
        xn, basen, H, W_qkv, 1, b_qkv, nullptr, nullptr, T, qkvbuf, M, 3 * H, H);
    // 3) attention
    attn_kernel<<<dim3(T / 64, B * 8), 256, 0, stream>>>(qkvbuf, attnb);
    // 4) output projection + residual(builder)
    gemm_kernel<<<dim3(M / 128, H / 64), 256, 0, stream>>>(
        attnb, attnb, BIG, W_o, 1, b_o, builder, nullptr, T, x1, M, H, H);
    // 5) norm2
    layernorm_kernel<<<M, 256, 0, stream>>>(x1, g2, beta2, xn, H);
    // 6) W_in projection
    gemm_kernel<<<dim3(M / 128, H / 64), 256, 0, stream>>>(
        xn, xn, BIG, W_in, 1, b_in, nullptr, nullptr, T, basen, M, H, H);
    // 7) U = hidden @ Bmat  (Bmat stored [K=H, N=S])
    gemm_kernel<<<dim3(M / 128, S / 64), 256, 0, stream>>>(
        basen, basen, BIG, Bmat, 0, nullptr, nullptr, nullptr, T, Ubuf, M, S, H);
    // 8) program modulation
    meanprog_kernel<<<1, 256, 0, stream>>>(pp, pm, T, P);
    mod_kernel<<<(B * H) / 256, 256, 0, stream>>>(pm, W_prog, b_prog, modb, P, H);
    // 9) sequential SSM scan
    scan_kernel<<<1, 256, 0, stream>>>(Amat, Ubuf, statesb, T);
    // 10) out = x1 + states @ Cmat + mod  (Cmat stored [K=S, N=H])
    gemm_kernel<<<dim3(M / 128, H / 64), 256, 0, stream>>>(
        statesb, statesb, BIG, Cmat, 0, nullptr, x1, modb, T, (float*)d_out,
        M, H, S);
}
